// SynchronAutoencoder_27659589386567
// MI455X (gfx1250) — matrix operand order repaired and checked by reading, not compiled, not run on hardware
//
#include <hip/hip_runtime.h>
#include <math.h>

// ---------------------------------------------------------------------------
// SynchronAutoencoder (3x SynchronGRU) for MI455X / gfx1250.
// Latency-bound problem: 3 x 512-step GRU scans + 3 x 15 sequential chain
// links. Strategy: persistent cooperative kernels, bf16 weights resident in
// LDS, v_wmma_f32_16x16x32_bf16 for the recurrent matvec, grid barrier per
// step, all chain-parallelizable work hoisted out of the sequential path.
// ---------------------------------------------------------------------------

#define HID    512      // GRU units
#define G3     1536     // 3*HID gate columns
#define NWG    32       // persistent workgroups (grid barrier members)
#define TPB    256      // 8 wave32 per WG
#define UPW    16       // hidden units owned per WG
#define CPW    48       // gate columns per WG (3 gates x 16)
#define NTILE  3        // 48 cols = 3 WMMA N=16 tiles
#define NLINK  15       // chain links per layer

typedef __attribute__((ext_vector_type(16))) __bf16 v16bf;
typedef __attribute__((ext_vector_type(8)))  float  v8f;

__device__ __forceinline__ unsigned short f2bf(float f) {
  unsigned x = __float_as_uint(f);
  x += 0x7FFFu + ((x >> 16) & 1u);          // round-to-nearest-even
  return (unsigned short)(x >> 16);
}
__device__ __forceinline__ float sigm(float x) { return 1.0f / (1.0f + __expf(-x)); }

// Monotonic-counter grid barrier across the NWG co-resident workgroups.
__device__ __forceinline__ void grid_barrier(unsigned* bar, unsigned* gen) {
  __syncthreads();
  if (threadIdx.x == 0) {
    __threadfence();                        // release h / scratch stores
    atomicAdd(bar, 1u);
    const unsigned target = (*gen + 1u) * NWG;
    while (__hip_atomic_load(bar, __ATOMIC_RELAXED, __HIP_MEMORY_SCOPE_AGENT) < target)
      __builtin_amdgcn_s_sleep(1);
    __threadfence();                        // acquire other WGs' stores
  }
  *gen += 1u;
  __syncthreads();
}

// ---------------------------------------------------------------------------
// Prep: rec (512x1536 f32, row-major [k][j]) -> per-WG bf16 A-fragment source
// aw[g][r][k] with r = gate*16+u, j = gate*512 + g*16 + u  (i.e. rec^T,
// grouped so each WG's 48x512 slice is contiguous for the LDS cache).
// ---------------------------------------------------------------------------
__global__ void prep_rect_bf16_kernel(const float* __restrict__ rec,
                                      unsigned short* __restrict__ aw) {
  const int idx = blockIdx.x * blockDim.x + threadIdx.x;
  if (idx >= NWG * CPW * HID) return;
  const int k = idx & (HID - 1);
  const int r = (idx >> 9) % CPW;
  const int g = idx / (CPW * HID);
  const int gate = r / UPW, u = r % UPW;
  const int j = gate * HID + g * UPW + u;
  aw[idx] = f2bf(rec[(size_t)k * G3 + j]);
}

// ---------------------------------------------------------------------------
// Main 512-step GRU scan (d_in = 1). One persistent launch per layer.
// Each step: all WGs rebuild bf16 h in LDS, run WMMA matvec from the
// LDS-resident weight slice, owner threads apply the gate math, publish h,
// grid-barrier.
// ---------------------------------------------------------------------------
__global__ __launch_bounds__(TPB) void gru_seq512_kernel(
    const float* __restrict__ in,           // layer input, accessor in[16*w + c]
    const float* __restrict__ k0,           // (1536)
    const float* __restrict__ b0,           // (2,1536): b_i then b_r
    const unsigned short* __restrict__ aw,  // prepped bf16 rec^T [NWG][48][512]
    float* __restrict__ hbuf,               // (512) f32 shared hidden state (pre-zeroed)
    float* __restrict__ yall,               // (16*512) output rows; row 0 written here
    unsigned* __restrict__ bar)
{
  __shared__ __align__(16) unsigned short wlds[CPW * HID];   // 48 KB weights
  __shared__ __align__(16) unsigned short hlds[HID];         // bf16 h
  __shared__ float acc[CPW];

  const int tid  = threadIdx.x, g = blockIdx.x;
  const int lane = tid & 31,    wave = tid >> 5;
  const int half = lane >> 4,   m = lane & 15;

  // Cache this WG's rec^T slice in LDS once; reused for all 512 steps.
  {
    const uint4* src = (const uint4*)(aw + (size_t)g * CPW * HID);
    uint4* dst = (uint4*)wlds;
    for (int i = tid; i < CPW * HID / 8; i += TPB) dst[i] = src[i];
  }

  float kz=0, kr=0, kn=0, biz=0, bir=0, bin=0, brz=0, brr=0, brn=0, hreg=0.f;
  if (tid < UPW) {
    const int u = g * UPW + tid;
    kz  = k0[u];        kr  = k0[HID + u];        kn  = k0[2*HID + u];
    biz = b0[u];        bir = b0[HID + u];        bin = b0[2*HID + u];
    brz = b0[G3 + u];   brr = b0[G3 + HID + u];   brn = b0[G3 + 2*HID + u];
  }
  __syncthreads();

  unsigned gen = 0;
  for (int t = 0; t < HID; ++t) {
    if (tid < CPW) acc[tid] = 0.f;
    hlds[tid]       = f2bf(hbuf[tid]);
    hlds[tid + 256] = f2bf(hbuf[tid + 256]);
    __syncthreads();

    // gh = h @ rec for this WG's 48 columns.
    // Wave w covers K-chunks {2w, 2w+1}; replicated-B trick: B[k,n] = h[k].
    for (int T = 0; T < NTILE; ++T) {
      v8f d = {};
      #pragma unroll
      for (int cc = 0; cc < 2; ++cc) {
        const int c = 2 * wave + cc;
        union { unsigned u[8]; v16bf v; } A, B;
        #pragma unroll
        for (int p = 0; p < 8; ++p) {
          // 16-bit A 16x32 fragment layout (ISA 7.12.2): K pair per VGPR.
          const int kb = 2*p + ((p >= 4) ? 8 : 0) + (half ? 8 : 0);
          A.u[p] = *(const unsigned*)&wlds[(T*16 + m) * HID + c*32 + kb];
          // 16-bit B 32x16: same K order as A (replicated).
          B.u[p] = *(const unsigned*)&hlds[c*32 + kb];
        }
        d = __builtin_amdgcn_wmma_f32_16x16x32_bf16(false, A.v, false, B.v,
                                                    (short)0, d, false, false);
      }
      // D layout: lane n=L%16 holds rows v + 8*half; all columns identical.
      if (m == 0) {
        #pragma unroll
        for (int v = 0; v < 8; ++v)
          atomicAdd(&acc[T*16 + v + 8*half], d[v]);      // ds_add_f32
      }
    }
    __syncthreads();

    if (tid < UPW) {                                      // gate math, owner units
      const float xt = in[16 * t];                        // column 0 of input
      const float z  = sigm(xt*kz + biz + acc[tid]        + brz);
      const float r  = sigm(xt*kr + bir + acc[UPW + tid]  + brr);
      const float hh = tanhf(xt*kn + bin + r * (acc[2*UPW + tid] + brn));
      hreg = z * hreg + (1.f - z) * hh;
      hbuf[g * UPW + tid] = hreg;
    }
    grid_barrier(bar, &gen);
  }
  if (tid < UPW) yall[g * UPW + tid] = hreg;              // y0 = row 0
}

// ---------------------------------------------------------------------------
// 15-link chain per layer. Phases A1-A3 precompute everything that does not
// depend on the sequential y_prev (gx1 from fixed x1, h1, gh2 = h1@rec+b_r).
// Phase B is the 15 sequential matvecs y_prev @ kernel_i + gate math.
// ---------------------------------------------------------------------------
__global__ __launch_bounds__(TPB) void gru_chain_kernel(
    const float* __restrict__ in,            // x1[w] = in[16*w + 1]
    const float* __restrict__ kc,            // (15,512,1536)
    const float* __restrict__ rkc,           // (15,512,1536)
    const float* __restrict__ bc,            // (15,2,1536)
    float* __restrict__ hb,                  // (1024) ping-pong y_prev; [0:512) = y0
    float* __restrict__ yall,                // rows 1..15 written
    float* __restrict__ gx1,                 // ws (15*1536)
    float* __restrict__ h1,                  // ws (15*512)
    float* __restrict__ gh2,                 // ws (15*1536)
    unsigned* __restrict__ bar)
{
  __shared__ float xl[HID];
  __shared__ float gx2[CPW];
  const int tid = threadIdx.x, g = blockIdx.x;
  const int gtid = g * TPB + tid;
  unsigned gen = 0;

  xl[tid]       = in[16 * tid + 1];
  xl[tid + 256] = in[16 * (tid + 256) + 1];
  __syncthreads();

  // A1: gx1[i][j] = x1 @ kernel_i + b_i   (parallel over 15*1536 columns)
  for (int task = gtid; task < NLINK * G3; task += NWG * TPB) {
    const int i = task / G3, j = task % G3;
    const float* K = kc + (size_t)i * HID * G3 + j;
    float s = bc[i * 2 * G3 + j];
    for (int k = 0; k < HID; ++k) {
      __builtin_prefetch(K + (size_t)(k + 24) * G3, 0, 1);   // global_prefetch_b8
      s += xl[k] * K[(size_t)k * G3];
    }
    gx1[task] = s;
  }
  grid_barrier(bar, &gen);

  // A2: h1[i][u] = first GRU step from h=0 (gh = b_r), elementwise.
  for (int task = gtid; task < NLINK * HID; task += NWG * TPB) {
    const int i = task / HID, u = task % HID;
    const float* br = bc + i * 2 * G3 + G3;
    const float z1  = sigm(gx1[i*G3 + u]            + br[u]);
    const float r1  = sigm(gx1[i*G3 + HID + u]      + br[HID + u]);
    const float hh1 = tanhf(gx1[i*G3 + 2*HID + u] + r1 * br[2*HID + u]);
    h1[task] = (1.f - z1) * hh1;
  }
  grid_barrier(bar, &gen);

  // A3: gh2[i][j] = h1_i @ rec_i + b_r   (parallel)
  for (int task = gtid; task < NLINK * G3; task += NWG * TPB) {
    const int i = task / G3, j = task % G3;
    const float* R  = rkc + (size_t)i * HID * G3 + j;
    const float* hi = h1 + i * HID;
    float s = bc[i * 2 * G3 + G3 + j];
    for (int k = 0; k < HID; ++k) {
      __builtin_prefetch(R + (size_t)(k + 24) * G3, 0, 1);
      s += hi[k] * R[(size_t)k * G3];
    }
    gh2[task] = s;
  }
  grid_barrier(bar, &gen);

  // B: sequential chain. Only gx2 = y_prev @ kernel_i remains on the path.
  for (int i = 0; i < NLINK; ++i) {
    const float* src = hb + (i & 1) * HID;        // y_prev
    float*       dst = hb + ((i + 1) & 1) * HID;  // y_i
    xl[tid]       = src[tid];
    xl[tid + 256] = src[tid + 256];
    __syncthreads();
    if (tid < CPW) {
      const int gate = tid / UPW, u = tid % UPW;
      const int j = gate * HID + g * UPW + u;
      const float* K = kc + (size_t)i * HID * G3 + j;
      float s = bc[i * 2 * G3 + j];
      for (int k = 0; k < HID; ++k) s += xl[k] * K[(size_t)k * G3];
      gx2[tid] = s;
    }
    __syncthreads();
    if (tid < UPW) {
      const int u = g * UPW + tid;
      const float z  = sigm(gx2[tid]         + gh2[i*G3 + u]);
      const float r  = sigm(gx2[UPW + tid]   + gh2[i*G3 + HID + u]);
      const float hh = tanhf(gx2[2*UPW + tid] + r * gh2[i*G3 + 2*HID + u]);
      const float y  = z * h1[i*HID + u] + (1.f - z) * hh;
      dst[u] = y;
      yall[(i + 1) * HID + u] = y;
    }
    grid_barrier(bar, &gen);
  }
}

__global__ void copy_out_kernel(const float* __restrict__ s, float* __restrict__ d, int n) {
  const int i = blockIdx.x * blockDim.x + threadIdx.x;
  if (i < n) d[i] = s[i];
}

// ---------------------------------------------------------------------------
// Host sequencing. Input order: x, then per layer l: k0, rk0, b0, kc, rkc, bc.
// The inter-layer reshape (1,512,16) is a flat reinterpretation, so every
// layer uses the same in[16*w + c] accessor and the final (16,512) buffer is
// copied flat to d_out.
// ---------------------------------------------------------------------------
extern "C" void kernel_launch(void* const* d_in, const int* in_sizes, int n_in,
                              void* d_out, int out_size, void* d_ws, size_t ws_size,
                              hipStream_t stream) {
  (void)in_sizes; (void)n_in; (void)ws_size;
  const float* x = (const float*)d_in[0];

  char* ws = (char*)d_ws;
  size_t off = 0;
  auto take = [&](size_t bytes) -> char* {
    char* p = ws + off;
    off = (off + bytes + 255) & ~(size_t)255;
    return p;
  };
  unsigned*       bars = (unsigned*)take(8 * sizeof(unsigned));
  unsigned short* aw[3];
  for (int l = 0; l < 3; ++l) aw[l] = (unsigned short*)take((size_t)NWG * CPW * HID * 2);
  float* hb  = (float*)take(2 * HID * sizeof(float));        // ping-pong hidden
  float* yA  = (float*)take(16 * HID * sizeof(float));
  float* yB  = (float*)take(16 * HID * sizeof(float));
  float* gx1 = (float*)take(NLINK * G3 * sizeof(float));
  float* h1  = (float*)take(NLINK * HID * sizeof(float));
  float* gh2 = (float*)take(NLINK * G3 * sizeof(float));

  hipMemsetAsync(bars, 0, 8 * sizeof(unsigned), stream);

  // Convert + transpose all three recurrent matrices to bf16 fragments.
  for (int l = 0; l < 3; ++l) {
    const float* rk0 = (const float*)d_in[1 + 6 * l + 1];
    prep_rect_bf16_kernel<<<(NWG * CPW * HID + TPB - 1) / TPB, TPB, 0, stream>>>(rk0, aw[l]);
  }

  const float* in = x;
  float* ybufs[3] = { yA, yB, yA };
  for (int l = 0; l < 3; ++l) {
    const float* k0  = (const float*)d_in[1 + 6 * l + 0];
    const float* b0  = (const float*)d_in[1 + 6 * l + 2];
    const float* kc  = (const float*)d_in[1 + 6 * l + 3];
    const float* rkc = (const float*)d_in[1 + 6 * l + 4];
    const float* bc  = (const float*)d_in[1 + 6 * l + 5];
    float* yout = ybufs[l];

    hipMemsetAsync(hb, 0, HID * sizeof(float), stream);      // h0 = 0
    gru_seq512_kernel<<<NWG, TPB, 0, stream>>>(in, k0, b0, aw[l], hb, yout, bars + l);
    gru_chain_kernel<<<NWG, TPB, 0, stream>>>(in, kc, rkc, bc, hb, yout,
                                              gx1, h1, gh2, bars + 3 + l);
    in = yout;
  }
  copy_out_kernel<<<(out_size + TPB - 1) / TPB, TPB, 0, stream>>>(ybufs[2], (float*)d_out, out_size);
}
